// TopologicalAttention_77266461655819
// MI455X (gfx1250) — matrix operand order repaired and checked by reading, not compiled, not run on hardware
//
#include <hip/hip_runtime.h>

// MI455X / gfx1250 wave32 WMMA TopologicalAttention.
// cvt(x) + transpose-cvt(W) -> topo proj -> 3x WMMA GEMM with BOTH tiles
// TDM-streamed (double-buffered, BK=64; Q pre-scaled; V written transposed)
// -> flash attention (WMMA + TDM K/V^T tiles, permlane16 softmax reductions)
// -> WMMA GEMM out projection (f32 + bias).

typedef _Float16 v16h __attribute__((ext_vector_type(16)));
typedef _Float16 v8h  __attribute__((ext_vector_type(8)));
typedef _Float16 v4h  __attribute__((ext_vector_type(4)));
typedef float    v8f  __attribute__((ext_vector_type(8)));
typedef float    v4f  __attribute__((ext_vector_type(4)));
typedef unsigned int v4u __attribute__((ext_vector_type(4)));
typedef int      v8i  __attribute__((ext_vector_type(8)));
typedef int      v4i  __attribute__((ext_vector_type(4)));

#define B_  4
#define S_  2048
#define D_  1024
#define H_  16
#define HD_ 64

#define KSCL 0.1803368801111204f   // (1/sqrt(64)) * log2(e)

static __device__ __forceinline__ v16h join16(v8h lo, v8h hi) {
  v16h r;
#pragma unroll
  for (int i = 0; i < 8; ++i) { r[i] = lo[i]; r[i + 8] = hi[i]; }
  return r;
}
static __device__ __forceinline__ v8f zero8() {
  v8f z;
#pragma unroll
  for (int i = 0; i < 8; ++i) z[i] = 0.f;
  return z;
}

// in-row (16-lane) xor butterflies. permlane16 keeps them in the VALU;
// selector nibbles: lane i reads row-index (i^mask).
#if __has_builtin(__builtin_amdgcn_permlane16)
static __device__ __forceinline__ float permx(float v, int lo, int hi) {
  int i = __builtin_bit_cast(int, v);
  i = __builtin_amdgcn_permlane16(i, i, lo, hi, false, false);
  return __builtin_bit_cast(float, i);
}
static __device__ __forceinline__ float rowmax16(float v) {
  v = fmaxf(v, permx(v, 0x67452301, (int)0xEFCDAB89));   // xor 1
  v = fmaxf(v, permx(v, 0x54761032, (int)0xDCFE98BA));   // xor 2
  v = fmaxf(v, permx(v, 0x32107654, (int)0xBA98FEDC));   // xor 4
  v = fmaxf(v, permx(v, (int)0xFEDCBA98, 0x76543210));   // xor 8
  return v;
}
static __device__ __forceinline__ float rowsum16(float v) {
  v += permx(v, 0x67452301, (int)0xEFCDAB89);
  v += permx(v, 0x54761032, (int)0xDCFE98BA);
  v += permx(v, 0x32107654, (int)0xBA98FEDC);
  v += permx(v, (int)0xFEDCBA98, 0x76543210);
  return v;
}
#else
static __device__ __forceinline__ float rowmax16(float v) {
  v = fmaxf(v, __shfl_xor(v, 1));
  v = fmaxf(v, __shfl_xor(v, 2));
  v = fmaxf(v, __shfl_xor(v, 4));
  v = fmaxf(v, __shfl_xor(v, 8));
  return v;
}
static __device__ __forceinline__ float rowsum16(float v) {
  v += __shfl_xor(v, 1);
  v += __shfl_xor(v, 2);
  v += __shfl_xor(v, 4);
  v += __shfl_xor(v, 8);
  return v;
}
#endif

// ------------------------------------------------------------------- TDM ----
// 2D tile load global->LDS via Tensor Data Mover (ISA 08_async_tensor §8).
// pad_int_code: dwords-before-pad (3->16dw, 4->32dw); pad_amt_code: pad dwords
// minus 1 (3 -> 4 dw = 8 halves) => pitch row+8 halves.
static __device__ __forceinline__ void tdm_load_2d(unsigned lds_off,
                                                   const void* gptr,
                                                   unsigned tile_d0,
                                                   unsigned tile_d1,
                                                   unsigned stride0,
                                                   unsigned pad_int_code,
                                                   unsigned pad_amt_code) {
  unsigned long long ga = (unsigned long long)(uintptr_t)gptr;
  v4u g0;
  g0[0] = 1u;                                        // count=1, user mode
  g0[1] = lds_off;                                   // lds_addr (bytes)
  g0[2] = (unsigned)ga;                              // global_addr[31:0]
  g0[3] = (unsigned)((ga >> 32) & 0x1FFFFFFu) | (2u << 30);  // [56:32] | type=2
  const unsigned td0 = 0x100000u, td1 = 0x100000u;   // huge dims: no OOB clip
  v8i g1;
  g1[0] = (int)((1u << 16) | (1u << 20) | (pad_int_code << 22) |
                (pad_amt_code << 25));               // data_size=2B, pad_en
  g1[1] = (int)((td0 & 0xFFFFu) << 16);              // atomic=0 | tdim0.lo
  g1[2] = (int)((td0 >> 16) | ((td1 & 0xFFFFu) << 16));
  g1[3] = (int)((td1 >> 16) | (tile_d0 << 16));      // tdim1.hi | tile_dim0
  g1[4] = (int)tile_d1;                              // tile_dim1 | tile_dim2=0
  g1[5] = (int)stride0;                              // tensor_dim0_stride lo32
  g1[6] = 0;
  g1[7] = 0;
  v4i gz;  gz[0] = gz[1] = gz[2] = gz[3] = 0;
#if defined(__clang_major__) && (__clang_major__ >= 23)
  v8i gz8;
#pragma unroll
  for (int i = 0; i < 8; ++i) gz8[i] = 0;
  __builtin_amdgcn_tensor_load_to_lds(g0, g1, gz, gz, gz8, 0);
#else
  __builtin_amdgcn_tensor_load_to_lds(g0, g1, gz, gz, 0);
#endif
}

// ---------------------------------------------------------------- convert ---
__global__ void cvt_f32_to_f16_v4(const float* __restrict__ src,
                                  _Float16* __restrict__ dst, int n4) {
  int i = blockIdx.x * blockDim.x + threadIdx.x;
  int stride = gridDim.x * blockDim.x;
  for (; i < n4; i += stride) {
    v4f v = ((const v4f*)src)[i];
    v4h h;
    h[0] = (_Float16)v[0]; h[1] = (_Float16)v[1];
    h[2] = (_Float16)v[2]; h[3] = (_Float16)v[3];
    ((v4h*)dst)[i] = h;
  }
}

// transpose + convert: W f32 [K][N] -> Wt f16 [N][K]  (64x64 LDS tiles)
__global__ __launch_bounds__(256)
void transpose_cvt(const float* __restrict__ src, _Float16* __restrict__ dst,
                   unsigned K, unsigned N) {
  __shared__ _Float16 t[64][72];
  const unsigned k0 = blockIdx.y * 64, n0 = blockIdx.x * 64;
  const unsigned tdx = threadIdx.x;
#pragma unroll
  for (int i = 0; i < 16; ++i) {
    unsigned idx = tdx + i * 256;
    unsigned r = idx >> 6, c = idx & 63;
    t[r][c] = (_Float16)src[(size_t)(k0 + r) * N + n0 + c];
  }
  __syncthreads();
#pragma unroll
  for (int i = 0; i < 16; ++i) {
    unsigned idx = tdx + i * 256;
    unsigned r = idx >> 6, c = idx & 63;
    dst[(size_t)(n0 + r) * K + k0 + c] = t[c][r];
  }
}

// ------------------------------------------------------- topo projection ----
__global__ void topo_proj(const float* __restrict__ tf,
                          const float* __restrict__ Wt,
                          const float* __restrict__ bt,
                          float* __restrict__ topo) {
  unsigned t = blockIdx.x * blockDim.x + threadIdx.x;
  unsigned n = t & (D_ - 1);
  unsigned b = t >> 10;
  float acc = bt[n];
  for (unsigned k = 0; k < D_; ++k)
    acc += tf[b * D_ + k] * Wt[(size_t)k * D_ + n];
  topo[b * D_ + n] = acc;
}

// ------------------------------------------------------------ WMMA GEMM -----
// C = A[M,K] f16 * Bt[N,K]^T f16 + bias (+ topo per batch if HAS_ADD).
// Both tiles TDM-streamed, double-buffered, BK=64 (pitch 72 halves).
// OUTMODE: 0 = f32 row-major, 1 = f16 row-major, 2 = f16 transposed-V
// [B][H][HD][S].  SCALE: fold KSCL (Q path).
template <bool HAS_ADD, int OUTMODE, bool SCALE>
__global__ __launch_bounds__(256)
void gemm_wmma(const _Float16* __restrict__ A, const _Float16* __restrict__ Bt,
               const float* __restrict__ bias, const float* __restrict__ addvec,
               _Float16* __restrict__ outF16, float* __restrict__ outF32,
               unsigned M, unsigned N, unsigned K) {
  constexpr int LDT = 72;
  __shared__ _Float16 As[2][128 * LDT];          // [m][k]
  __shared__ _Float16 Bs[2][128 * LDT];          // [n][k]

  const unsigned tid = threadIdx.x;
  const unsigned lane = tid & 31, wave = tid >> 5;
  const unsigned wm = wave >> 2, wn = wave & 3;
  const unsigned g = lane >> 4, l16 = lane & 15;
  const unsigned blockM = blockIdx.x * 128, blockN = blockIdx.y * 128;
  const bool w0 = (__builtin_amdgcn_readfirstlane(wave) == 0);

  v8f acc[4][2];
#pragma unroll
  for (int i = 0; i < 4; ++i)
#pragma unroll
    for (int j = 0; j < 2; ++j) acc[i][j] = zero8();

  const _Float16* Abase = A + (size_t)blockM * K;
  const _Float16* Bbase = Bt + (size_t)blockN * K;

  if (w0) {
    tdm_load_2d((unsigned)(uintptr_t)&As[0][0], Abase, 64, 128, K, 4, 3);
    tdm_load_2d((unsigned)(uintptr_t)&Bs[0][0], Bbase, 64, 128, K, 4, 3);
    __builtin_amdgcn_s_wait_tensorcnt(0);
  }
  __syncthreads();

  const unsigned nIter = K / 64;
  for (unsigned it = 0; it < nIter; ++it) {
    const int cur = it & 1, nxt = cur ^ 1;
    const bool more = (it + 1 < nIter);
    if (more && w0) {
      tdm_load_2d((unsigned)(uintptr_t)&As[nxt][0], Abase + (it + 1) * 64,
                  64, 128, K, 4, 3);
      tdm_load_2d((unsigned)(uintptr_t)&Bs[nxt][0], Bbase + (it + 1) * 64,
                  64, 128, K, 4, 3);
    }

#pragma unroll
    for (int kk = 0; kk < 2; ++kk) {
      v16h afrag[4];
#pragma unroll
      for (int mt = 0; mt < 4; ++mt) {
        const _Float16* p =
            &As[cur][(wm * 64 + mt * 16 + l16) * LDT + kk * 32 + g * 8];
        afrag[mt] = join16(*(const v8h*)p, *(const v8h*)(p + 16));
      }
      v16h bfrag[2];
#pragma unroll
      for (int nt = 0; nt < 2; ++nt) {
        const _Float16* p =
            &Bs[cur][(wn * 32 + nt * 16 + l16) * LDT + kk * 32 + g * 8];
        bfrag[nt] = join16(*(const v8h*)p, *(const v8h*)(p + 16));
      }
#pragma unroll
      for (int mt = 0; mt < 4; ++mt)
#pragma unroll
        for (int nt = 0; nt < 2; ++nt)
          acc[mt][nt] = __builtin_amdgcn_wmma_f32_16x16x32_f16(
              false, afrag[mt], false, bfrag[nt], (short)0, acc[mt][nt],
              false, false);
    }

    if (w0) __builtin_amdgcn_s_wait_tensorcnt(0);
    __syncthreads();
  }

  // epilogue (batch index uniform per block: 128 | 2048)
  const unsigned bIdx = blockM >> 11;
  const unsigned mBase = blockM + wm * 64 + 8 * g;
#pragma unroll
  for (int nt = 0; nt < 2; ++nt) {
    const unsigned n = blockN + wn * 32 + nt * 16 + l16;
    float colAdd = bias[n];
    if constexpr (HAS_ADD) colAdd += addvec[bIdx * N + n];
#pragma unroll
    for (int mt = 0; mt < 4; ++mt) {
      if constexpr (OUTMODE == 2) {
        // V^T [B][H][HD][S]: lane's 8 rows are consecutive s -> one b128 store
        const unsigned hh = n >> 6, dd = n & 63;
        const unsigned s = (mBase & (S_ - 1)) + mt * 16;
        v8h pk;
#pragma unroll
        for (int r = 0; r < 8; ++r)
          pk[r] = (_Float16)(acc[mt][nt][r] + colAdd);
        *(v8h*)(outF16 + (((size_t)(bIdx * H_ + hh) * HD_ + dd) * S_) + s) = pk;
      } else if constexpr (OUTMODE == 1) {
        _Float16* p = outF16 + (size_t)(mBase + mt * 16) * N + n;
#pragma unroll
        for (int r = 0; r < 8; ++r) {
          float v = acc[mt][nt][r] + colAdd;
          if constexpr (SCALE) v *= KSCL;
          p[(size_t)r * N] = (_Float16)v;
        }
      } else {
        float* p = outF32 + (size_t)(mBase + mt * 16) * N + n;
#pragma unroll
        for (int r = 0; r < 8; ++r) p[(size_t)r * N] = acc[mt][nt][r] + colAdd;
      }
    }
  }
}

// ------------------------------------------------------- flash attention ----
// Grid (S/64, B*H), 128 thr = 4 waves, 16 q rows/wave. K (row-major) and V^T
// tiles streamed by TDM into double-buffered padded LDS (pitch 72 halves).
__global__ __launch_bounds__(128)
void flash_attn(const _Float16* __restrict__ Q, const _Float16* __restrict__ Kk,
                const _Float16* __restrict__ Vt, _Float16* __restrict__ O) {
  constexpr int LDK = 72;
  __shared__ _Float16 Ks[2][64 * LDK];           // [kv][d]
  __shared__ _Float16 Vs[2][64 * LDK];           // [d][kv]  (from V^T)
  __shared__ _Float16 Ps[4][16 * LDK];           // per-wave P strip [q][kv]

  const unsigned tid = threadIdx.x;
  const unsigned lane = tid & 31, wave = tid >> 5;
  const unsigned g = lane >> 4, l16 = lane & 15;
  const unsigned q0 = blockIdx.x * 64;
  const unsigned b = blockIdx.y >> 4, h = blockIdx.y & 15;
  const bool w0 = (__builtin_amdgcn_readfirstlane(wave) == 0);

  // Q already scaled by KSCL at projection time (exp2 domain)
  const unsigned qrow = q0 + wave * 16 + l16;
  const _Float16* qp = Q + ((size_t)(b * S_ + qrow) * D_) + h * HD_;
  v16h qfrag[2];
#pragma unroll
  for (int c = 0; c < 2; ++c)
    qfrag[c] = join16(*(const v8h*)(qp + c * 32 + g * 8),
                      *(const v8h*)(qp + c * 32 + 16 + g * 8));

  float Mrow[8], Lrow[8];
  v8f Oacc[4];
#pragma unroll
  for (int r = 0; r < 8; ++r) { Mrow[r] = -1e30f; Lrow[r] = 0.f; }
#pragma unroll
  for (int dt = 0; dt < 4; ++dt) Oacc[dt] = zero8();

  _Float16* Pw = &Ps[wave][0];
  const _Float16* kBase = Kk + ((size_t)(b * S_) * D_) + h * HD_;   // + kv*D
  const _Float16* vBase = Vt + ((size_t)(b * H_ + h) * HD_) * S_;   // + d*S+kv

  if (w0) {
    tdm_load_2d((unsigned)(uintptr_t)&Ks[0][0], kBase, 64, 64, D_, 4, 3);
    tdm_load_2d((unsigned)(uintptr_t)&Vs[0][0], vBase, 64, 64, S_, 4, 3);
  }

  const unsigned nIter = S_ / 64;
  for (unsigned it = 0; it < nIter; ++it) {
    const int cur = it & 1, nxt = cur ^ 1;
    if (w0) {
      if (it + 1 < nIter) {
        const unsigned kv = (it + 1) * 64;
        tdm_load_2d((unsigned)(uintptr_t)&Ks[nxt][0], kBase + (size_t)kv * D_,
                    64, 64, D_, 4, 3);
        tdm_load_2d((unsigned)(uintptr_t)&Vs[nxt][0], vBase + kv,
                    64, 64, S_, 4, 3);
        __builtin_amdgcn_s_wait_tensorcnt(2);   // current tile's 2 loads done
      } else {
        __builtin_amdgcn_s_wait_tensorcnt(0);
      }
    }
    __syncthreads();

    // S' = Qscaled K^T
    v8f sacc[4];
#pragma unroll
    for (int kt = 0; kt < 4; ++kt) {
      v8f a = zero8();
#pragma unroll
      for (int c = 0; c < 2; ++c) {
        const _Float16* p = &Ks[cur][(kt * 16 + l16) * LDK + c * 32 + g * 8];
        v16h bf = join16(*(const v8h*)p, *(const v8h*)(p + 16));
        a = __builtin_amdgcn_wmma_f32_16x16x32_f16(false, qfrag[c], false, bf,
                                                   (short)0, a, false, false);
      }
      sacc[kt] = a;
    }

    // online softmax (exp2 domain); rows live in 16-lane half-groups
    float mnew[8], alpha[8];
#pragma unroll
    for (int r = 0; r < 8; ++r) {
      float mx = fmaxf(fmaxf(sacc[0][r], sacc[1][r]),
                       fmaxf(sacc[2][r], sacc[3][r]));
      mx = rowmax16(mx);
      mnew[r] = fmaxf(Mrow[r], mx);
      alpha[r] = __builtin_exp2f(Mrow[r] - mnew[r]);
      Mrow[r] = mnew[r];
    }
#pragma unroll
    for (int r = 0; r < 8; ++r) {
      float rs = 0.f;
#pragma unroll
      for (int kt = 0; kt < 4; ++kt) {
        float p = __builtin_exp2f(sacc[kt][r] - mnew[r]);
        rs += p;
        Pw[(r + 8 * g) * LDK + kt * 16 + l16] = (_Float16)p;
      }
      Lrow[r] = Lrow[r] * alpha[r] + rowsum16(rs);
    }
#pragma unroll
    for (int dt = 0; dt < 4; ++dt)
#pragma unroll
      for (int r = 0; r < 8; ++r) Oacc[dt][r] *= alpha[r];

    // O += P @ V
#pragma unroll
    for (int c = 0; c < 2; ++c) {
      const _Float16* pp = &Pw[l16 * LDK + c * 32 + g * 8];
      v16h pfrag = join16(*(const v8h*)pp, *(const v8h*)(pp + 16));
#pragma unroll
      for (int dt = 0; dt < 4; ++dt) {
        const _Float16* vp =
            &Vs[cur][(dt * 16 + l16) * LDK + c * 32 + g * 8];
        v16h vf = join16(*(const v8h*)vp, *(const v8h*)(vp + 16));
        Oacc[dt] = __builtin_amdgcn_wmma_f32_16x16x32_f16(
            false, pfrag, false, vf, (short)0, Oacc[dt], false, false);
      }
    }
    __syncthreads();   // reads of buf[cur] done before TDM overwrites it
  }

#pragma unroll
  for (int r = 0; r < 8; ++r) {
    const unsigned qq = q0 + wave * 16 + r + 8 * g;
    _Float16* op = O + ((size_t)(b * S_ + qq) * D_) + h * HD_ + l16;
    const float inv = 1.f / Lrow[r];
#pragma unroll
    for (int dt = 0; dt < 4; ++dt) op[dt * 16] = (_Float16)(Oacc[dt][r] * inv);
  }
}

// --------------------------------------------------------------- launch -----
extern "C" void kernel_launch(void* const* d_in, const int* in_sizes, int n_in,
                              void* d_out, int out_size, void* d_ws,
                              size_t ws_size, hipStream_t stream) {
  const float* x  = (const float*)d_in[0];
  const float* tf = (const float*)d_in[1];
  const float* Wq = (const float*)d_in[2];
  const float* bq = (const float*)d_in[3];
  const float* Wk = (const float*)d_in[4];
  const float* bk = (const float*)d_in[5];
  const float* Wv = (const float*)d_in[6];
  const float* bv = (const float*)d_in[7];
  const float* Wt = (const float*)d_in[8];
  const float* bt = (const float*)d_in[9];
  const float* Wo = (const float*)d_in[10];
  const float* bo = (const float*)d_in[11];
  float* out = (float*)d_out;

  char* ws = (char*)d_ws;
  size_t off = 0;
  auto alloc = [&](size_t bytes) -> char* {
    char* p = ws + off;
    off = (off + bytes + 255) & ~(size_t)255;
    return p;
  };
  const size_t MSD = (size_t)B_ * S_ * D_;
  const size_t DD  = (size_t)D_ * D_;
  _Float16* xh  = (_Float16*)alloc(MSD * 2);
  _Float16* wqt = (_Float16*)alloc(DD * 2);      // W^T f16 [N][K]
  _Float16* wkt = (_Float16*)alloc(DD * 2);
  _Float16* wvt = (_Float16*)alloc(DD * 2);
  _Float16* wot = (_Float16*)alloc(DD * 2);
  float*    topo = (float*)alloc((size_t)B_ * D_ * 4);
  _Float16* Qh  = (_Float16*)alloc(MSD * 2);
  _Float16* Kh  = (_Float16*)alloc(MSD * 2);
  _Float16* Vth = (_Float16*)alloc(MSD * 2);     // V^T [B][H][HD][S]
  _Float16* Ah  = (_Float16*)alloc(MSD * 2);

  cvt_f32_to_f16_v4<<<1024, 256, 0, stream>>>(x, xh, (int)(MSD / 4));
  dim3 gt(D_ / 64, D_ / 64);
  transpose_cvt<<<gt, 256, 0, stream>>>(Wq, wqt, D_, D_);
  transpose_cvt<<<gt, 256, 0, stream>>>(Wk, wkt, D_, D_);
  transpose_cvt<<<gt, 256, 0, stream>>>(Wv, wvt, D_, D_);
  transpose_cvt<<<gt, 256, 0, stream>>>(Wo, wot, D_, D_);

  topo_proj<<<(B_ * D_) / 256, 256, 0, stream>>>(tf, Wt, bt, topo);

  dim3 gg(B_ * S_ / 128, D_ / 128);
  // Q: f16 row-major, pre-scaled by KSCL
  gemm_wmma<false, 1, true><<<gg, 256, 0, stream>>>(xh, wqt, bq, nullptr, Qh,
                                                    nullptr, B_ * S_, D_, D_);
  // K: f16 row-major + topo bias
  gemm_wmma<true, 1, false><<<gg, 256, 0, stream>>>(xh, wkt, bk, topo, Kh,
                                                    nullptr, B_ * S_, D_, D_);
  // V: f16 transposed [B][H][HD][S]
  gemm_wmma<false, 2, false><<<gg, 256, 0, stream>>>(xh, wvt, bv, nullptr, Vth,
                                                     nullptr, B_ * S_, D_, D_);

  flash_attn<<<dim3(S_ / 64, B_ * H_), 128, 0, stream>>>(Qh, Kh, Vth, Ah);

  // output projection: f32 + bias
  gemm_wmma<false, 0, false><<<gg, 256, 0, stream>>>(Ah, wot, bo, nullptr,
                                                     nullptr, out, B_ * S_, D_,
                                                     D_);
}
